// GNNTransformerEncoder_23270132809807
// MI455X (gfx1250) — matrix operand order repaired and checked by reading, not compiled, not run on hardware
//
#include <hip/hip_runtime.h>
#include <stdint.h>

// ---------------------------------------------------------------------------
// Problem constants (match reference)
// ---------------------------------------------------------------------------
#define NNODES 50000
#define NEDGES 1600000
#define INCH   128      // IN == H*C
#define NH     4
#define NC     32
#define NL     3
#define NG     64
#define NOUT   64
#define MT_PER_BLOCK 4  // M-tiles per block (per wave), reusing B fragments

typedef __attribute__((ext_vector_type(16))) __bf16 v16bf;
typedef __attribute__((ext_vector_type(8)))  float  v8f;

union Frag { v16bf v; uint32_t u[8]; };

// ---------------------------------------------------------------------------
// Device helpers
// ---------------------------------------------------------------------------
__device__ __forceinline__ void atomicAddF(float* p, float v) {
  __hip_atomic_fetch_add(p, v, __ATOMIC_RELAXED, __HIP_MEMORY_SCOPE_AGENT);
}

// float atomic max via signed-int max (positives) / unsigned min (negatives).
// Initialize target to 0xFF800000 (-inf).
__device__ __forceinline__ void atomicMaxF(float* addr, float val) {
  if (!(val < 0.0f)) atomicMax((int*)addr, __float_as_int(val));
  else               atomicMin((unsigned int*)addr, __float_as_uint(val));
}

// ---------------------------------------------------------------------------
// f32 -> bf16 convert (layer-0 x)
// ---------------------------------------------------------------------------
__global__ void cvt_f32_bf16(const float* __restrict__ in, __bf16* __restrict__ out, int n) {
  int i = blockIdx.x * blockDim.x + threadIdx.x;
  if (i < n) out[i] = (__bf16)in[i];
}

// ---------------------------------------------------------------------------
// Pack one 128x128 f32 weight into bf16 WMMA B-fragment order:
//   out[nt(8)][kt(4)][lane(32)][i(16)]
// 32x16 bf16 B tile: lanes 0-15 hold K=0..15 (col = lane), lanes 16-31 hold
// K=16..31 (col = lane-16); fragment element i <-> K = kh*16 + i.
// ---------------------------------------------------------------------------
__global__ void pack_w_bf16(const float* __restrict__ W, __bf16* __restrict__ out) {
  int j = blockIdx.x * blockDim.x + threadIdx.x;   // 0 .. 16383
  if (j >= INCH * INCH) return;
  int i    = j & 15;
  int lane = (j >> 4) & 31;
  int kt   = (j >> 9) & 3;
  int nt   = j >> 11;
  int col  = lane & 15;
  int kh   = lane >> 4;
  out[j] = (__bf16)W[(size_t)(kt * 32 + kh * 16 + i) * INCH + nt * 16 + col];
}

__global__ void zero_f32(float* __restrict__ p, int n) {
  int i = blockIdx.x * blockDim.x + threadIdx.x;
  if (i < n) p[i] = 0.0f;
}

__global__ void init_softmax(float* __restrict__ amax, float* __restrict__ asum, int n) {
  int i = blockIdx.x * blockDim.x + threadIdx.x;
  if (i < n) { ((unsigned int*)amax)[i] = 0xFF800000u; asum[i] = 0.0f; }
}

// ---------------------------------------------------------------------------
// WMMA GEMM:  Out[M,128] = A[M,128] (bf16) * W (packed bf16) + bias
// block = 256 threads (8 waves); wave w -> N-tile w; block -> MT_PER_BLOCK
// M-tiles, reusing the 4 preloaded B fragments across all of them.
// ---------------------------------------------------------------------------
__global__ __launch_bounds__(256) void gemm_bf16_wmma(
    const __bf16* __restrict__ A, const __bf16* __restrict__ Wpk,
    const float* __restrict__ bias, float* __restrict__ Out, int M)
{
  const int lane = threadIdx.x & 31;
  const int wave = threadIdx.x >> 5;   // n-tile
  const int n0   = wave * 16;
  const int col  = lane & 15;
  const int kh   = lane >> 4;

  // Preload B fragments for the 4 K-tiles: contiguous, coalesced b128 loads.
  Frag b[4];
#pragma unroll
  for (int kt = 0; kt < 4; ++kt) {
    const uint32_t* bp =
        (const uint32_t*)(Wpk + ((size_t)(wave * 4 + kt) * 32 + lane) * 16);
#pragma unroll
    for (int d = 0; d < 8; ++d) b[kt].u[d] = bp[d];
  }

  const float bs = bias[n0 + col];
  const int rb = kh * 8;

#pragma unroll
  for (int mt = 0; mt < MT_PER_BLOCK; ++mt) {
    const int m0 = (blockIdx.x * MT_PER_BLOCK + mt) * 16;
    if (m0 >= M) return;

    v8f acc = {};
    const size_t rowBase = (size_t)(m0 + col) * INCH;
#pragma unroll
    for (int kt = 0; kt < 4; ++kt) {
      // 16x32 bf16 A: lane holds row (lane&15); K halves {0..15}/{16..31}.
      Frag a;
      const uint32_t* p0 = (const uint32_t*)(A + rowBase + kt * 32 + kh * 16);
      const uint32_t* p1 = (const uint32_t*)(A + rowBase + kt * 32 + kh * 16 + 8);
      a.u[0] = p0[0]; a.u[1] = p0[1]; a.u[2] = p0[2]; a.u[3] = p0[3];
      a.u[4] = p1[0]; a.u[5] = p1[1]; a.u[6] = p1[2]; a.u[7] = p1[3];
      acc = __builtin_amdgcn_wmma_f32_16x16x32_bf16(
          false, a.v, false, b[kt].v, (short)0, acc, false, false);
    }

    // f32 C/D layout: VGPR r -> row r (lanes 0-15) / row 8+r (lanes 16-31).
#pragma unroll
    for (int r = 0; r < 8; ++r)
      Out[(size_t)(m0 + rb + r) * INCH + n0 + col] = acc[r] + bs;
  }
}

// ---------------------------------------------------------------------------
// Edge pass 1: alpha[e,h] = (q[dst]·k[src])_h * scale ; segment-max into amax
// One wave32 per edge; lane loads float4 (4 of 128); 8-lane groups = one head.
// ---------------------------------------------------------------------------
__global__ __launch_bounds__(256) void edge_alpha(
    const int* __restrict__ src, const int* __restrict__ dst,
    const float* __restrict__ q, const float* __restrict__ k,
    float* __restrict__ alpha, float* __restrict__ amax)
{
  int e    = (int)((blockIdx.x * blockDim.x + threadIdx.x) >> 5);
  int lane = threadIdx.x & 31;
  if (e >= NEDGES) return;
  __builtin_prefetch(src + e + 4096, 0, 0);
  __builtin_prefetch(dst + e + 4096, 0, 0);
  int s = src[e], d = dst[e];
  float4 qq = ((const float4*)(q + (size_t)d * INCH))[lane];
  float4 kk = ((const float4*)(k + (size_t)s * INCH))[lane];
  float p = qq.x * kk.x + qq.y * kk.y + qq.z * kk.z + qq.w * kk.w;
  p += __shfl_xor(p, 1, 32);
  p += __shfl_xor(p, 2, 32);
  p += __shfl_xor(p, 4, 32);
  if ((lane & 7) == 0) {
    int h = lane >> 3;
    float al = p * 0.17677669529663687f;   // 1/sqrt(32)
    alpha[(size_t)e * NH + h] = al;
    atomicMaxF(&amax[(size_t)d * NH + h], al);
  }
}

// ---------------------------------------------------------------------------
// Edge pass 2: a = exp(alpha - amax[dst]) ; segment-sum into asum
// ---------------------------------------------------------------------------
__global__ void edge_exp(const int* __restrict__ dst, float* __restrict__ alpha,
                         const float* __restrict__ amax, float* __restrict__ asum)
{
  int i = blockIdx.x * blockDim.x + threadIdx.x;
  if (i >= NEDGES * NH) return;
  int e = i >> 2, h = i & 3;
  int d = dst[e];
  float a = expf(alpha[i] - amax[(size_t)d * NH + h]);
  alpha[i] = a;
  atomicAddF(&asum[(size_t)d * NH + h], a);
}

// ---------------------------------------------------------------------------
// Edge pass 3: agg[dst,c] += (a / asum[dst]) * v[src,c]   (agg holds skip)
// 128 consecutive threads per edge -> coalesced v reads / agg atomics (L2-resident).
// ---------------------------------------------------------------------------
__global__ void edge_scatter(const int* __restrict__ src, const int* __restrict__ dst,
                             const float* __restrict__ alpha, const float* __restrict__ asum,
                             const float* __restrict__ v, float* __restrict__ agg)
{
  int i = blockIdx.x * blockDim.x + threadIdx.x;
  if (i >= NEDGES * INCH) return;
  int e = i >> 7, c = i & 127, h = c >> 5;
  int d = dst[e], s = src[e];
  float w = alpha[(size_t)e * NH + h] / (asum[(size_t)d * NH + h] + 1e-16f);
  atomicAddF(&agg[(size_t)d * INCH + c], w * v[(size_t)s * INCH + c]);
}

// ---------------------------------------------------------------------------
// x = relu(agg)   (also emit bf16 copy for next layer's GEMMs)
// ---------------------------------------------------------------------------
__global__ void relu_cvt(const float* __restrict__ agg, float* __restrict__ xf,
                         __bf16* __restrict__ xb, int n)
{
  int i = blockIdx.x * blockDim.x + threadIdx.x;
  if (i >= n) return;
  float t = agg[i];
  t = t > 0.0f ? t : 0.0f;
  xf[i] = t;
  xb[i] = (__bf16)t;
}

// ---------------------------------------------------------------------------
// Global mean pool (sums + counts)
// ---------------------------------------------------------------------------
__global__ void pool_sum(const float* __restrict__ xf, const int* __restrict__ batch,
                         float* __restrict__ pooled, float* __restrict__ cnt, int n)
{
  int i = blockIdx.x * blockDim.x + threadIdx.x;
  if (i >= n) return;
  int node = i >> 7, c = i & 127;
  int g = batch[node];
  atomicAddF(&pooled[(size_t)g * INCH + c], xf[i]);
  if (c == 0) atomicAddF(&cnt[g], 1.0f);
}

// ---------------------------------------------------------------------------
// out[g,o] = mean(pooled[g]) · fcW[:,o] + fcb[o]   (tiny; fp32 scalar)
// ---------------------------------------------------------------------------
__global__ void final_fc(const float* __restrict__ pooled, const float* __restrict__ cnt,
                         const float* __restrict__ fcW, const float* __restrict__ fcb,
                         float* __restrict__ out)
{
  int i = blockIdx.x * blockDim.x + threadIdx.x;
  if (i >= NG * NOUT) return;
  int g = i >> 6, o = i & 63;
  float inv = 1.0f / fmaxf(cnt[g], 1.0f);
  float acc = fcb[o];
#pragma unroll 4
  for (int c = 0; c < INCH; ++c)
    acc += pooled[(size_t)g * INCH + c] * inv * fcW[(size_t)c * NOUT + o];
  out[i] = acc;
}

// ---------------------------------------------------------------------------
// Host launcher
// ---------------------------------------------------------------------------
static inline char* carve(char*& p, size_t bytes) {
  char* r = p;
  p += (bytes + 255) & ~(size_t)255;
  return r;
}

extern "C" void kernel_launch(void* const* d_in, const int* in_sizes, int n_in,
                              void* d_out, int out_size, void* d_ws, size_t ws_size,
                              hipStream_t stream) {
  const float* x_in  = (const float*)d_in[0];
  const int*   eidx  = (const int*)d_in[1];
  const int*   batch = (const int*)d_in[2];
  const float* Wmats[4] = { (const float*)d_in[3], (const float*)d_in[5],
                            (const float*)d_in[7], (const float*)d_in[9] };
  const float* bmats[4] = { (const float*)d_in[4], (const float*)d_in[6],
                            (const float*)d_in[8], (const float*)d_in[10] };
  const float* fcW = (const float*)d_in[11], *fcb = (const float*)d_in[12];
  float* out = (float*)d_out;

  const int* src = eidx;            // edge_index[0]
  const int* dst = eidx + NEDGES;   // edge_index[1]

  char* p = (char*)d_ws;
  float*  q     = (float*)carve(p, (size_t)NNODES * INCH * 4);
  float*  kbuf  = (float*)carve(p, (size_t)NNODES * INCH * 4);
  float*  vbuf  = (float*)carve(p, (size_t)NNODES * INCH * 4);
  float*  agg   = (float*)carve(p, (size_t)NNODES * INCH * 4);
  float*  xf    = (float*)carve(p, (size_t)NNODES * INCH * 4);
  __bf16* xb    = (__bf16*)carve(p, (size_t)NNODES * INCH * 2);
  float*  alpha = (float*)carve(p, (size_t)NEDGES * NH * 4);
  float*  amax  = (float*)carve(p, (size_t)NNODES * NH * 4);
  float*  asum  = (float*)carve(p, (size_t)NNODES * NH * 4);
  __bf16* wpk   = (__bf16*)carve(p, (size_t)4 * NL * INCH * INCH * 2); // [mat][l][packed]
  float*  pooled = (float*)carve(p, (size_t)NG * INCH * 4);
  float*  cnt    = (float*)carve(p, (size_t)NG * 4);

  // Pack + convert all 12 weight matrices into WMMA B-fragment order.
  const int WM = INCH * INCH;                          // 16384
  for (int m = 0; m < 4; ++m)
    for (int l = 0; l < NL; ++l)
      pack_w_bf16<<<WM / 256, 256, 0, stream>>>(
          Wmats[m] + (size_t)l * WM, wpk + ((size_t)m * NL + l) * WM);

  const int XN = NNODES * INCH;                        // 6.4M
  cvt_f32_bf16<<<(XN + 255) / 256, 256, 0, stream>>>(x_in, xb, XN);

  zero_f32<<<(NG * INCH + 255) / 256, 256, 0, stream>>>(pooled, NG * INCH);
  zero_f32<<<1, 256, 0, stream>>>(cnt, NG);

  const int SM = NNODES * NH;                          // 200K
  const int EW = NEDGES / 8;                           // wave-per-edge blocks
  const int MB = (NNODES / 16 + MT_PER_BLOCK - 1) / MT_PER_BLOCK; // GEMM blocks
  float* gout[4] = { q, kbuf, vbuf, agg };             // skip -> agg

  for (int l = 0; l < NL; ++l) {
    init_softmax<<<(SM + 255) / 256, 256, 0, stream>>>(amax, asum, SM);

    for (int m = 0; m < 4; ++m)
      gemm_bf16_wmma<<<MB, 256, 0, stream>>>(
          xb, wpk + ((size_t)m * NL + l) * WM, bmats[m] + l * INCH, gout[m], NNODES);

    edge_alpha<<<EW, 256, 0, stream>>>(src, dst, q, kbuf, alpha, amax);
    edge_exp<<<(NEDGES * NH + 255) / 256, 256, 0, stream>>>(dst, alpha, amax, asum);
    edge_scatter<<<(int)(((size_t)NEDGES * INCH + 255) / 256), 256, 0, stream>>>(
        src, dst, alpha, asum, vbuf, agg);

    relu_cvt<<<(XN + 255) / 256, 256, 0, stream>>>(agg, xf, xb, XN);
  }

  pool_sum<<<(XN + 255) / 256, 256, 0, stream>>>(xf, batch, pooled, cnt, XN);
  final_fc<<<(NG * NOUT + 255) / 256, 256, 0, stream>>>(pooled, cnt, fcW, fcb, out);
}
